// GCNModelVAE_XA_2173253451797
// MI455X (gfx1250) — compile-verified, matrix operand order checked by reading, not run on hardware
//
#include <hip/hip_runtime.h>
#include <hip/hip_bf16.h>

// ---------------------------------------------------------------------------
// GCN-VAE forward for MI455X (gfx1250, wave32, WMMA).
// Dense GEMMs: tiled bf16 WMMA (v_wmma_f32_16x16x32_bf16), 128x128 block
// tile, double-buffered LDS, async global->LDS copies for the A tile
// (global_load_async_to_lds_b128 + s_wait_asynccnt). Fused bias/BN/ReLU.
// SpMM (segment_sum over edges): global_atomic_add_f32 scatter.
// ---------------------------------------------------------------------------

typedef __bf16 bf16_t;
typedef __attribute__((ext_vector_type(16))) __bf16 v16bf;
typedef __attribute__((ext_vector_type(8)))  __bf16 v8bf;
typedef __attribute__((ext_vector_type(4)))  __bf16 v4bf;
typedef __attribute__((ext_vector_type(8)))  float  v8f;

// Problem constants (match reference)
#define NN   10000
#define EE   320000
#define DD   3000
#define HH1  1024
#define HH2  128
#define FF1  256
#define FF2  512
#define FF3  1024
#define FF4  2048

#if defined(__AMDGCN__) && __has_builtin(__builtin_amdgcn_global_load_async_to_lds_b128)
#define USE_ASYNC_LDS 1
#endif

#ifdef USE_ASYNC_LDS
typedef int v4i_t __attribute__((vector_size(16)));
typedef __attribute__((address_space(1))) v4i_t as1v4i;
typedef __attribute__((address_space(3))) v4i_t as3v4i;
__device__ __forceinline__ void async_copy16(const bf16_t* g, bf16_t* l) {
    __builtin_amdgcn_global_load_async_to_lds_b128(
        (as1v4i*)(unsigned long long)g,
        (as3v4i*)(unsigned)(unsigned long long)l, 0, 0);
}
__device__ __forceinline__ void wait_async0() {
#if __has_builtin(__builtin_amdgcn_s_wait_asynccnt)
    __builtin_amdgcn_s_wait_asynccnt(0);
#else
    asm volatile("s_wait_asynccnt 0x0" ::: "memory");
#endif
}
#endif

// ---------------------------------------------------------------------------
// f32 -> bf16 conversion, 4-wide (all sizes here are multiples of 4)
// ---------------------------------------------------------------------------
__global__ void cvt_f32_bf16(const float* __restrict__ in, bf16_t* __restrict__ out, long n4) {
    long i = (long)blockIdx.x * blockDim.x + threadIdx.x;
    long stride = (long)gridDim.x * blockDim.x;
    for (; i < n4; i += stride) {
        float4 v = ((const float4*)in)[i];
        v4bf o; o[0] = (bf16_t)v.x; o[1] = (bf16_t)v.y; o[2] = (bf16_t)v.z; o[3] = (bf16_t)v.w;
        ((v4bf*)out)[i] = o;
    }
}

__global__ void relu_cvt_f32_bf16(const float* __restrict__ in, bf16_t* __restrict__ out, long n4) {
    long i = (long)blockIdx.x * blockDim.x + threadIdx.x;
    long stride = (long)gridDim.x * blockDim.x;
    for (; i < n4; i += stride) {
        float4 v = ((const float4*)in)[i];
        v4bf o;
        o[0] = (bf16_t)fmaxf(v.x, 0.f); o[1] = (bf16_t)fmaxf(v.y, 0.f);
        o[2] = (bf16_t)fmaxf(v.z, 0.f); o[3] = (bf16_t)fmaxf(v.w, 0.f);
        ((v4bf*)out)[i] = o;
    }
}

// out[c*rows + r] = (bf16)in[r*cols + c], cols = 1<<lgc
__global__ void transpose_cvt(const float* __restrict__ in, bf16_t* __restrict__ out,
                              int rows, int lgc) {
    long total = (long)rows << lgc;
    const int cmask = (1 << lgc) - 1;
    long i = (long)blockIdx.x * blockDim.x + threadIdx.x;
    long stride = (long)gridDim.x * blockDim.x;
    for (; i < total; i += stride) {
        int r = (int)(i >> lgc);
        int c = (int)(i & cmask);
        out[(long)c * rows + r] = (bf16_t)in[i];
    }
}

// ---------------------------------------------------------------------------
// SpMM scatter: out[dst[e]*F + f] += w[e] * h[src[e]*F + f]
// F = 1<<lgF; 4 floats per thread via global_atomic_add_f32. out pre-zeroed.
// ---------------------------------------------------------------------------
__global__ void spmm_atomic(const float* __restrict__ h, const float* __restrict__ ew,
                            const int* __restrict__ src, const int* __restrict__ dst,
                            float* __restrict__ out, int E, int lgF) {
    const int lgq = lgF - 2;             // feature quads per edge
    const int qmask = (1 << lgq) - 1;
    const int F = 1 << lgF;
    long total = (long)E << lgq;
    long i = (long)blockIdx.x * blockDim.x + threadIdx.x;
    long stride = (long)gridDim.x * blockDim.x;
    for (; i < total; i += stride) {
        int e  = (int)(i >> lgq);
        int f4 = (int)(i & qmask) << 2;
        float w = ew[e];
        int s = src[e], d = dst[e];
        const float4 hv = *(const float4*)(h + (size_t)s * F + f4);
        float* o = out + (size_t)d * F + f4;
        atomicAdd(o + 0, w * hv.x);
        atomicAdd(o + 1, w * hv.y);
        atomicAdd(o + 2, w * hv.z);
        atomicAdd(o + 3, w * hv.w);
    }
}

// ---------------------------------------------------------------------------
// Tiled WMMA GEMM: C(MxN) = A(MxK,bf16,rowmaj) @ B(KxN,bf16,rowmaj)
// Block tile 128x128, k-step 32. 256 threads = 8 waves (wave32) in 4(M)x2(N);
// each wave owns 32x64 = 2x4 v_wmma_f32_16x16x32_bf16 tiles.
// Interior blocks: unguarded, double-buffered LDS, one barrier per k-step,
// A tile via async global->LDS copies when available.
// Edge blocks: guarded single-buffer path.
// Epilogue: y = relu_opt(((c + bias) * inv * gamma) + beta).
// ---------------------------------------------------------------------------
template <typename OutT>
__global__ __launch_bounds__(256)
void gemm_wmma_bf16(const bf16_t* __restrict__ A, const bf16_t* __restrict__ B,
                    OutT* __restrict__ C, int M, int N, int K,
                    const float* __restrict__ bias,
                    const float* __restrict__ gamma,
                    const float* __restrict__ beta,
                    float inv, int relu) {
    __shared__ bf16_t As[2][128 * 32];   // [row][k], row stride 32
    __shared__ bf16_t Bs[2][128 * 32];   // [col][k] (transposed in LDS)

    const int tid   = threadIdx.x;
    const int lane  = tid & 31;
    const int wave  = tid >> 5;
    const int warpM = wave & 3;          // 4 row groups of 32 rows
    const int warpN = wave >> 2;         // 2 col groups of 64 cols
    const int half  = lane >> 4;
    const int l15   = lane & 15;

    const int m0 = blockIdx.y * 128;
    const int n0 = blockIdx.x * 128;

    v8f acc[2][4];
#pragma unroll
    for (int a = 0; a < 2; ++a)
#pragma unroll
        for (int b = 0; b < 4; ++b)
            acc[a][b] = (v8f){0.f, 0.f, 0.f, 0.f, 0.f, 0.f, 0.f, 0.f};

    const int nkt = (K + 31) >> 5;

    // staging coords: 2 chunks of 8 bf16 (16B) per thread per tile, per matrix
    int ar[2], ak[2], bk[2], bn[2];
#pragma unroll
    for (int i = 0; i < 2; ++i) {
        const int c = tid + i * 256;     // 0..511
        ar[i] = c >> 2;                  // A: 0..127 rows
        ak[i] = (c & 3) << 3;            // A: k offset 0,8,16,24
        bk[i] = c >> 4;                  // B: 0..31 k rows
        bn[i] = (c & 15) << 3;           // B: n offset 0..120
    }

    // fragment loads per ISA layout (lanes 0-15: K 0..7 & 16..23;
    // lanes 16-31: K 8..15 & 24..31) + 8 WMMAs
    auto compute = [&](int buf) {
        v16bf af[2], bfr[4];
#pragma unroll
        for (int tm = 0; tm < 2; ++tm) {
            const bf16_t* p = &As[buf][(warpM * 32 + tm * 16 + l15) * 32];
            v8bf lo = *(const v8bf*)(p + half * 8);
            v8bf hi = *(const v8bf*)(p + 16 + half * 8);
            af[tm] = __builtin_shufflevector(lo, hi, 0, 1, 2, 3, 4, 5, 6, 7,
                                             8, 9, 10, 11, 12, 13, 14, 15);
        }
#pragma unroll
        for (int tn = 0; tn < 4; ++tn) {
            const bf16_t* p = &Bs[buf][(warpN * 64 + tn * 16 + l15) * 32];
            v8bf lo = *(const v8bf*)(p + half * 8);
            v8bf hi = *(const v8bf*)(p + 16 + half * 8);
            bfr[tn] = __builtin_shufflevector(lo, hi, 0, 1, 2, 3, 4, 5, 6, 7,
                                              8, 9, 10, 11, 12, 13, 14, 15);
        }
#pragma unroll
        for (int tm = 0; tm < 2; ++tm)
#pragma unroll
            for (int tn = 0; tn < 4; ++tn)
                acc[tm][tn] = __builtin_amdgcn_wmma_f32_16x16x32_bf16(
                    false, af[tm], false, bfr[tn], (short)0, acc[tm][tn],
                    false, false);
    };

    if ((m0 + 128 <= M) && (n0 + 128 <= N)) {
        // ------------------ interior fast path ------------------
        v8bf bst[2];
        auto loadB = [&](int kt, v8bf* o2) {
            const int k0 = kt << 5;
            if (k0 + 32 <= K) {          // uniform scalar branch
#pragma unroll
                for (int i = 0; i < 2; ++i)
                    o2[i] = *(const v8bf*)(B + (size_t)(k0 + bk[i]) * N + (n0 + bn[i]));
            } else {                     // K-remainder tile
#pragma unroll
                for (int i = 0; i < 2; ++i) {
                    const int gk = k0 + bk[i];
                    if (gk < K) {
                        o2[i] = *(const v8bf*)(B + (size_t)gk * N + (n0 + bn[i]));
                    } else {
#pragma unroll
                        for (int j = 0; j < 8; ++j) o2[i][j] = (bf16_t)0.0f;
                    }
                }
            }
        };
        auto storeB = [&](const v8bf* i2, int buf) {
#pragma unroll
            for (int i = 0; i < 2; ++i)
#pragma unroll
                for (int j = 0; j < 8; ++j)
                    Bs[buf][(bn[i] + j) * 32 + bk[i]] = i2[i][j];
        };

#ifdef USE_ASYNC_LDS
        auto fillA = [&](int kt, int buf) {
            const int k0 = kt << 5;
            if (k0 + 32 <= K) {
#pragma unroll
                for (int i = 0; i < 2; ++i)
                    async_copy16(A + (size_t)(m0 + ar[i]) * K + (k0 + ak[i]),
                                 &As[buf][ar[i] * 32 + ak[i]]);
            } else {
#pragma unroll
                for (int i = 0; i < 2; ++i)
#pragma unroll
                    for (int j = 0; j < 8; ++j) {
                        const int kk = k0 + ak[i] + j;
                        As[buf][ar[i] * 32 + ak[i] + j] =
                            (kk < K) ? A[(size_t)(m0 + ar[i]) * K + kk] : (bf16_t)0.0f;
                    }
            }
        };
        fillA(0, 0);
        loadB(0, bst);
        for (int kt = 0; kt < nkt; ++kt) {
            const int cur = kt & 1;
            storeB(bst, cur);
            const bool more = (kt + 1 < nkt);
            v8bf bnx[2];
            if (more) loadB(kt + 1, bnx);
            wait_async0();               // A(kt) landed in LDS
            __syncthreads();             // B stores visible; all waves synced
            if (more) fillA(kt + 1, cur ^ 1);
            compute(cur);
            if (more) { bst[0] = bnx[0]; bst[1] = bnx[1]; }
        }
#else
        v8bf ast[2];
        auto loadA = [&](int kt, v8bf* o2) {
            const int k0 = kt << 5;
            if (k0 + 32 <= K) {
#pragma unroll
                for (int i = 0; i < 2; ++i)
                    o2[i] = *(const v8bf*)(A + (size_t)(m0 + ar[i]) * K + (k0 + ak[i]));
            } else {
#pragma unroll
                for (int i = 0; i < 2; ++i)
#pragma unroll
                    for (int j = 0; j < 8; ++j) {
                        const int kk = k0 + ak[i] + j;
                        o2[i][j] = (kk < K) ? A[(size_t)(m0 + ar[i]) * K + kk] : (bf16_t)0.0f;
                    }
            }
        };
        auto storeA = [&](const v8bf* i2, int buf) {
#pragma unroll
            for (int i = 0; i < 2; ++i)
                *(v8bf*)&As[buf][ar[i] * 32 + ak[i]] = i2[i];
        };
        loadA(0, ast);
        loadB(0, bst);
        for (int kt = 0; kt < nkt; ++kt) {
            const int cur = kt & 1;
            storeA(ast, cur);
            storeB(bst, cur);
            const bool more = (kt + 1 < nkt);
            v8bf anx[2], bnx[2];
            if (more) { loadA(kt + 1, anx); loadB(kt + 1, bnx); }
            __syncthreads();
            compute(cur);
            if (more) {
                ast[0] = anx[0]; ast[1] = anx[1];
                bst[0] = bnx[0]; bst[1] = bnx[1];
            }
        }
#endif
    } else {
        // ------------------ guarded edge path (single buffer) ------------------
        for (int kt = 0; kt < nkt; ++kt) {
            const int k0 = kt << 5;
#pragma unroll
            for (int i = 0; i < 2; ++i) {
                const int grow = m0 + ar[i];
#pragma unroll
                for (int j = 0; j < 8; ++j) {
                    const int kk = k0 + ak[i] + j;
                    As[0][ar[i] * 32 + ak[i] + j] =
                        (grow < M && kk < K) ? A[(size_t)grow * K + kk] : (bf16_t)0.0f;
                }
                const int gk = k0 + bk[i];
#pragma unroll
                for (int j = 0; j < 8; ++j) {
                    const int gn = n0 + bn[i] + j;
                    Bs[0][(bn[i] + j) * 32 + bk[i]] =
                        (gk < K && gn < N) ? B[(size_t)gk * N + gn] : (bf16_t)0.0f;
                }
            }
            __syncthreads();
            compute(0);
            __syncthreads();
        }
    }

    // ---- epilogue + store. C/D layout: lane n = l15, m = half*8 + r ----
#pragma unroll
    for (int tm = 0; tm < 2; ++tm) {
#pragma unroll
        for (int tn = 0; tn < 4; ++tn) {
            const int col = n0 + warpN * 64 + tn * 16 + l15;
#pragma unroll
            for (int r = 0; r < 8; ++r) {
                const int row = m0 + warpM * 32 + tm * 16 + half * 8 + r;
                if (row < M && col < N) {
                    float v = acc[tm][tn][r];
                    if (bias)  v += bias[col];
                    if (gamma) v = v * inv * gamma[col] + beta[col];
                    if (relu)  v = fmaxf(v, 0.0f);
                    C[(size_t)row * N + col] = (OutT)v;
                }
            }
        }
    }
}

// ---------------------------------------------------------------------------
// Host side
// ---------------------------------------------------------------------------
static inline unsigned cdiv(long a, long b) { return (unsigned)((a + b - 1) / b); }

static void launch_cvt(const float* in, bf16_t* out, long n, hipStream_t s) {
    long n4 = n >> 2;
    unsigned g = cdiv(n4, 256); if (g > 16384) g = 16384;
    cvt_f32_bf16<<<g, 256, 0, s>>>(in, out, n4);
}

template <typename OutT>
static void launch_gemm(const bf16_t* A, const bf16_t* B, OutT* C, int M, int N, int K,
                        const float* bias, const float* gamma, const float* beta,
                        float inv, int relu, hipStream_t s) {
    dim3 grid(cdiv(N, 128), cdiv(M, 128));
    gemm_wmma_bf16<OutT><<<grid, 256, 0, s>>>(A, B, C, M, N, K, bias, gamma, beta, inv, relu);
}

extern "C" void kernel_launch(void* const* d_in, const int* in_sizes, int n_in,
                              void* d_out, int out_size, void* d_ws, size_t ws_size,
                              hipStream_t stream) {
    (void)in_sizes; (void)n_in; (void)out_size; (void)ws_size;

    // ---- inputs (setup_inputs dict order) ----
    const float* x      = (const float*)d_in[0];
    const float* edge_w = (const float*)d_in[1];
    const float* W1     = (const float*)d_in[2];
    const float* W2     = (const float*)d_in[3];
    const float* W3     = (const float*)d_in[4];
    const float* fcw[5] = {(const float*)d_in[5],  (const float*)d_in[9],
                           (const float*)d_in[13], (const float*)d_in[17],
                           (const float*)d_in[21]};
    const float* fcb[5] = {(const float*)d_in[6],  (const float*)d_in[10],
                           (const float*)d_in[14], (const float*)d_in[18],
                           (const float*)d_in[22]};
    const float* bng[4] = {(const float*)d_in[7],  (const float*)d_in[11],
                           (const float*)d_in[15], (const float*)d_in[19]};
    const float* bnb[4] = {(const float*)d_in[8],  (const float*)d_in[12],
                           (const float*)d_in[16], (const float*)d_in[20]};
    const int* edge_src = (const int*)d_in[23];
    const int* edge_dst = (const int*)d_in[24];

    // ---- outputs (concatenated: adj_rec, mu, logvar, z, x_rec) ----
    float* out      = (float*)d_out;
    float* adj_rec  = out;
    float* mu       = out + (size_t)NN * NN;
    float* logvar   = mu  + (size_t)NN * HH2;
    float* z        = logvar + (size_t)NN * HH2;
    float* x_rec    = z + (size_t)NN * HH2;

    // ---- workspace bump allocator (256B aligned) ----
    char* w = (char*)d_ws;
    size_t off = 0;
    auto alloc = [&](size_t bytes) -> void* {
        void* p = w + off;
        off += (bytes + 255) & ~(size_t)255;
        return p;
    };
    bf16_t* xbf   = (bf16_t*)alloc((size_t)NN * DD * 2);   // later reused as h1pre (f32)
    bf16_t* w1bf  = (bf16_t*)alloc((size_t)DD * HH1 * 2);
    bf16_t* w2bf  = (bf16_t*)alloc((size_t)HH1 * HH2 * 2);
    bf16_t* w3bf  = (bf16_t*)alloc((size_t)HH1 * HH2 * 2);
    bf16_t* f1w   = (bf16_t*)alloc((size_t)HH2 * FF1 * 2);
    bf16_t* f2w   = (bf16_t*)alloc((size_t)FF1 * FF2 * 2);
    bf16_t* f3w   = (bf16_t*)alloc((size_t)FF2 * FF3 * 2);
    bf16_t* f4w   = (bf16_t*)alloc((size_t)FF3 * FF4 * 2);
    bf16_t* f5w   = (bf16_t*)alloc((size_t)FF4 * DD * 2);
    float*  xw1   = (float*) alloc((size_t)NN * HH1 * 4);  // later reused as o4 (bf16)
    bf16_t* h1bf  = (bf16_t*)alloc((size_t)NN * HH1 * 2);
    float*  hw2   = (float*) alloc((size_t)NN * HH2 * 4);
    float*  hw3   = (float*) alloc((size_t)NN * HH2 * 4);
    bf16_t* zbf   = (bf16_t*)alloc((size_t)NN * HH2 * 2);
    bf16_t* ztbf  = (bf16_t*)alloc((size_t)NN * HH2 * 2);
    bf16_t* o1    = (bf16_t*)alloc((size_t)NN * FF1 * 2);
    bf16_t* o2    = (bf16_t*)alloc((size_t)NN * FF2 * 2);
    bf16_t* o3    = (bf16_t*)alloc((size_t)NN * FF3 * 2);
    // aliases (lifetimes do not overlap):
    float*  h1pre = (float*)xbf;     // 40.96MB into 60MB region
    bf16_t* o4    = (bf16_t*)xw1;    // 40.96MB region

    const float inv = 0.999995000037f; // 1/sqrt(1 + 1e-5)

    // 1) convert x and all weights to bf16
    launch_cvt(x,  xbf,  (long)NN * DD,  stream);
    launch_cvt(W1, w1bf, (long)DD * HH1, stream);
    launch_cvt(W2, w2bf, (long)HH1 * HH2, stream);
    launch_cvt(W3, w3bf, (long)HH1 * HH2, stream);
    launch_cvt(fcw[0], f1w, (long)HH2 * FF1, stream);
    launch_cvt(fcw[1], f2w, (long)FF1 * FF2, stream);
    launch_cvt(fcw[2], f3w, (long)FF2 * FF3, stream);
    launch_cvt(fcw[3], f4w, (long)FF3 * FF4, stream);
    launch_cvt(fcw[4], f5w, (long)FF4 * DD,  stream);

    // 2) xw1 = x @ W1
    launch_gemm<float>(xbf, w1bf, xw1, NN, HH1, DD, nullptr, nullptr, nullptr, 1.f, 0, stream);

    // 3) h1pre = spmm(xw1)
    (void)hipMemsetAsync(h1pre, 0, (size_t)NN * HH1 * 4, stream);
    {
        long total = (long)EE * (HH1 / 4);
        unsigned g = cdiv(total, 256); if (g > 32768) g = 32768;
        spmm_atomic<<<g, 256, 0, stream>>>(xw1, edge_w, edge_src, edge_dst, h1pre, EE, 10);
    }

    // 4) h1 = relu(h1pre) -> bf16
    {
        long n4 = ((long)NN * HH1) >> 2;
        unsigned g = cdiv(n4, 256); if (g > 16384) g = 16384;
        relu_cvt_f32_bf16<<<g, 256, 0, stream>>>(h1pre, h1bf, n4);
    }

    // 5) hw2 = h1 @ W2 ; hw3 = h1 @ W3
    launch_gemm<float>(h1bf, w2bf, hw2, NN, HH2, HH1, nullptr, nullptr, nullptr, 1.f, 0, stream);
    launch_gemm<float>(h1bf, w3bf, hw3, NN, HH2, HH1, nullptr, nullptr, nullptr, 1.f, 0, stream);

    // 6) mu = spmm(hw2) ; logvar = spmm(hw3)
    (void)hipMemsetAsync(mu,     0, (size_t)NN * HH2 * 4, stream);
    (void)hipMemsetAsync(logvar, 0, (size_t)NN * HH2 * 4, stream);
    {
        long total = (long)EE * (HH2 / 4);
        unsigned g = cdiv(total, 256); if (g > 32768) g = 32768;
        spmm_atomic<<<g, 256, 0, stream>>>(hw2, edge_w, edge_src, edge_dst, mu,     EE, 7);
        spmm_atomic<<<g, 256, 0, stream>>>(hw3, edge_w, edge_src, edge_dst, logvar, EE, 7);
    }

    // 7) z = mu (copy into output slot)
    (void)hipMemcpyAsync(z, mu, (size_t)NN * HH2 * 4, hipMemcpyDeviceToDevice, stream);

    // 8) zbf = bf16(mu) ; ztbf = bf16(mu)^T  (128 x 10000)
    launch_cvt(mu, zbf, (long)NN * HH2, stream);
    {
        long n = (long)NN * HH2;
        unsigned g = cdiv(n, 256); if (g > 16384) g = 16384;
        transpose_cvt<<<g, 256, 0, stream>>>(mu, ztbf, NN, 7);
    }

    // 9) adj_rec = z @ z^T   (10000 x 10000, K=128)
    launch_gemm<float>(zbf, ztbf, adj_rec, NN, NN, HH2, nullptr, nullptr, nullptr, 1.f, 0, stream);

    // 10-13) FC stack with fused bias + eval-BN + ReLU, bf16 chaining
    launch_gemm<bf16_t>(zbf, f1w, o1, NN, FF1, HH2, fcb[0], bng[0], bnb[0], inv, 1, stream);
    launch_gemm<bf16_t>(o1,  f2w, o2, NN, FF2, FF1, fcb[1], bng[1], bnb[1], inv, 1, stream);
    launch_gemm<bf16_t>(o2,  f3w, o3, NN, FF3, FF2, fcb[2], bng[2], bnb[2], inv, 1, stream);
    launch_gemm<bf16_t>(o3,  f4w, o4, NN, FF4, FF3, fcb[3], bng[3], bnb[3], inv, 1, stream);
    // fc5: bias only, f32 out
    launch_gemm<float>(o4, f5w, x_rec, NN, DD, FF4, fcb[4], nullptr, nullptr, 1.f, 0, stream);
}
